// Gemma4Attention_40922448397009
// MI455X (gfx1250) — matrix operand order repaired and checked by reading, not compiled, not run on hardware
//
#include <hip/hip_runtime.h>
#include <hip/hip_bf16.h>

// ---------------- problem constants (match reference) ----------------
#define B_      2
#define S_      2048
#define HIDDEN_ 2560
#define H_      8
#define KV_     4
#define HD_     256
#define GROUPS_ 2
#define MAXSEQ_ 4096
#define NQKV_   (H_ * HD_ + 2 * KV_ * HD_)   // 4096 qkv output cols
#define LN_THETA 13.815510557964274f         // ln(1e6)

typedef __attribute__((ext_vector_type(16))) __bf16 v16bf;
typedef __attribute__((ext_vector_type(2)))  __bf16 v2bf;
typedef __attribute__((ext_vector_type(8)))  float  v8f;

union B16V { v16bf v; uint4 q[2]; unsigned short u[16]; };

__device__ __forceinline__ unsigned short f2b(float f) {
    unsigned int u = __float_as_uint(f);
    unsigned int r = u + 0x7FFFu + ((u >> 16) & 1u);   // round-to-nearest-even
    return (unsigned short)(r >> 16);
}
__device__ __forceinline__ unsigned int pack2(float a, float b) {
#if __has_builtin(__builtin_amdgcn_cvt_pk_bf16_f32)
    union { v2bf v; unsigned int u; } cv;
    cv.v = __builtin_amdgcn_cvt_pk_bf16_f32(a, b);
    return cv.u;
#else
    return (unsigned int)f2b(a) | ((unsigned int)f2b(b) << 16);
#endif
}

__device__ __forceinline__ v8f wmma_bf16(const B16V& a, const B16V& b, v8f c) {
    return __builtin_amdgcn_wmma_f32_16x16x32_bf16(false, a.v, false, b.v,
                                                   (short)0, c, false, false);
}

// ---------------------------------------------------------------------
// Tensor Data Mover: 1-D bf16 tile, global -> LDS (wave-uniform).
// ---------------------------------------------------------------------
#if __has_builtin(__builtin_amdgcn_tensor_load_to_lds)
#define HAVE_TDM 1
typedef unsigned int u32x4 __attribute__((ext_vector_type(4)));
typedef int          i32x4 __attribute__((ext_vector_type(4)));
typedef int          i32x8 __attribute__((ext_vector_type(8)));

__device__ __forceinline__ void tdm_load_1d_bf16(unsigned lds_off,
                                                 const void* gptr,
                                                 unsigned nelem)
{
    const unsigned long long ga = (unsigned long long)(size_t)gptr;
    u32x4 g0;
    g0[0] = 1u;                                             // count=1 (user D#)
    g0[1] = lds_off;                                        // lds_addr (bytes)
    g0[2] = (unsigned)(ga & 0xFFFFFFFFu);                   // global_addr lo
    g0[3] = (unsigned)((ga >> 32) & 0x01FFFFFFu)            // global_addr[56:32]
            | (2u << 30);                                   // type = 2 (image)
    i32x8 g1;
    g1[0] = (int)(1u << 16);                                // data_size = 1 -> 2B
    g1[1] = (int)((nelem & 0xFFFFu) << 16);                 // tensor_dim0 lo16
    g1[2] = (int)(nelem >> 16);                             // tensor_dim0 hi16
    g1[3] = (int)((nelem & 0xFFFFu) << 16);                 // tile_dim0 (<=65535)
    g1[4] = 0;                                              // tile_dim1/2 unused
    g1[5] = (int)nelem;                                     // tensor_dim0_stride
    g1[6] = 0;
    g1[7] = 0;
    i32x4 z4 = {0, 0, 0, 0};
#if defined(__clang_major__) && __clang_major__ >= 23
    i32x8 z8 = {0, 0, 0, 0, 0, 0, 0, 0};
    __builtin_amdgcn_tensor_load_to_lds(g0, g1, z4, z4, z8, 0);
#else
    __builtin_amdgcn_tensor_load_to_lds(g0, g1, z4, z4, 0);
#endif
}
#else
#define HAVE_TDM 0
#endif

// =====================================================================
// One-shot conversion kernels (run once per launch; weights/x then stay
// bf16 + L2-resident for the WMMA kernels).
// =====================================================================

// x [4096][2560] f32 -> bf16 (same layout)
__global__ __launch_bounds__(256)
void cvt_x_kernel(const float* __restrict__ x, unsigned short* __restrict__ xbf)
{
    const size_t i8 = ((size_t)blockIdx.x * 256 + threadIdx.x) * 8;
    const float4 f0 = *(const float4*)(x + i8);
    const float4 f1 = *(const float4*)(x + i8 + 4);
    uint4 pk;
    pk.x = pack2(f0.x, f0.y);
    pk.y = pack2(f0.z, f0.w);
    pk.z = pack2(f1.x, f1.y);
    pk.w = pack2(f1.z, f1.w);
    *(uint4*)(xbf + i8) = pk;
}

// QKV weights -> wT [4096 cols][2560 k] bf16 (transposed, col-major GEMM B)
// cols 0-2047: w_q, 2048-3071: w_k, 3072-4095: w_v
__global__ __launch_bounds__(256)
void cvt_w_kernel(const float* __restrict__ w_q,
                  const float* __restrict__ w_k,
                  const float* __restrict__ w_v,
                  unsigned short* __restrict__ wT)
{
    const int k0  = blockIdx.x * 32;                 // 80 chunks
    const int col = blockIdx.y * 256 + threadIdx.x;  // 4096 cols
    const float* src; int rowlen, c;
    if (col < 2048)      { src = w_q; rowlen = 2048; c = col; }
    else if (col < 3072) { src = w_k; rowlen = 1024; c = col - 2048; }
    else                 { src = w_v; rowlen = 1024; c = col - 3072; }
    const float* p = src + (size_t)k0 * rowlen + c;
    unsigned short* dst = wT + (size_t)col * HIDDEN_ + k0;
    #pragma unroll
    for (int jj = 0; jj < 4; ++jj) {
        uint4 pk;
        pk.x = pack2(p[(size_t)(jj*8+0)*rowlen], p[(size_t)(jj*8+1)*rowlen]);
        pk.y = pack2(p[(size_t)(jj*8+2)*rowlen], p[(size_t)(jj*8+3)*rowlen]);
        pk.z = pack2(p[(size_t)(jj*8+4)*rowlen], p[(size_t)(jj*8+5)*rowlen]);
        pk.w = pack2(p[(size_t)(jj*8+6)*rowlen], p[(size_t)(jj*8+7)*rowlen]);
        *(uint4*)(dst + jj * 8) = pk;
    }
}

// w_o [2048][2560] -> woT [2560 cols][2048 k] bf16
__global__ __launch_bounds__(256)
void cvt_wo_kernel(const float* __restrict__ w_o, unsigned short* __restrict__ woT)
{
    const int k0  = blockIdx.x * 32;                 // 64 chunks
    const int col = blockIdx.y * 256 + threadIdx.x;  // 2560 cols
    const float* p = w_o + (size_t)k0 * HIDDEN_ + col;
    unsigned short* dst = woT + (size_t)col * (H_ * HD_) + k0;
    #pragma unroll
    for (int jj = 0; jj < 4; ++jj) {
        uint4 pk;
        pk.x = pack2(p[(size_t)(jj*8+0)*HIDDEN_], p[(size_t)(jj*8+1)*HIDDEN_]);
        pk.y = pack2(p[(size_t)(jj*8+2)*HIDDEN_], p[(size_t)(jj*8+3)*HIDDEN_]);
        pk.z = pack2(p[(size_t)(jj*8+4)*HIDDEN_], p[(size_t)(jj*8+5)*HIDDEN_]);
        pk.w = pack2(p[(size_t)(jj*8+6)*HIDDEN_], p[(size_t)(jj*8+7)*HIDDEN_]);
        *(uint4*)(dst + jj * 8) = pk;
    }
}

// =====================================================================
// Kernel 1: QKV GEMM, pure WMMA main loop (A/B direct from bf16 global,
// no LDS staging, no barriers), fused RMSNorm + RoPE epilogue.
// grid = (M/32, 16 heads): heads 0-7 -> Q, 8-11 -> K, 12-15 -> V.
// =====================================================================
__global__ __launch_bounds__(256)
void qkv_kernel(const unsigned short* __restrict__ xbf,
                const unsigned short* __restrict__ wT,
                const float* __restrict__ qn,
                const float* __restrict__ kn,
                const int*   __restrict__ sp,
                unsigned short* __restrict__ qbf,
                unsigned short* __restrict__ kfull,
                unsigned short* __restrict__ vfull)
{
    const int tid  = threadIdx.x;
    const int lane = tid & 31;
    const int wv   = tid >> 5;          // 8 waves
    const int l16  = lane & 15;
    const int hf   = lane >> 4;

    const int m0   = blockIdx.x * 32;   // token tile
    const int head = blockIdx.y;        // 0..15; col base = head*256 in wT
    const int start_pos = sp[0];
    const int mode = (head < 8) ? 0 : (head < 12 ? 1 : 2);
    const int hh   = (head < 8) ? head : (head < 12 ? head - 8 : head - 12);

    __shared__ float ldsC[32 * 256];    // 32 KB epilogue tile
    __shared__ float ldsRed[32 * 8];
    __shared__ float ldsRms[32];

    const int rb = wv & 1;              // row block (16 rows)
    const int cb = wv >> 1;             // col quad  (64 cols)

    v8f acc[4];
    {
        v8f z = {0.f,0.f,0.f,0.f,0.f,0.f,0.f,0.f};
        #pragma unroll
        for (int i = 0; i < 4; ++i) acc[i] = z;
    }

    const unsigned short* arow = xbf + (size_t)(m0 + rb * 16 + l16) * HIDDEN_;
    const unsigned short* bcol[4];
    #pragma unroll
    for (int i = 0; i < 4; ++i)
        bcol[i] = wT + (size_t)(head * HD_ + (cb * 4 + i) * 16 + l16) * HIDDEN_;

    for (int k0 = 0; k0 < HIDDEN_; k0 += 32) {
        B16V a;
        a.q[0] = *(const uint4*)(arow + k0 + hf * 8);
        a.q[1] = *(const uint4*)(arow + k0 + 16 + hf * 8);
        #pragma unroll
        for (int i = 0; i < 4; ++i) {
            B16V bm;
            bm.q[0] = *(const uint4*)(bcol[i] + k0 + hf * 8);
            bm.q[1] = *(const uint4*)(bcol[i] + k0 + 16 + hf * 8);
            acc[i] = wmma_bf16(a, bm, acc[i]);
        }
    }

    // ---------------- epilogue: C -> LDS, RMSNorm, RoPE, store ----------------
    #pragma unroll
    for (int i = 0; i < 4; ++i) {
        const int col = (cb * 4 + i) * 16 + l16;
        const int rbase = rb * 16 + hf * 8;
        #pragma unroll
        for (int j = 0; j < 8; ++j) ldsC[(rbase + j) * 256 + col] = acc[i][j];
    }
    __syncthreads();
    {
        const int r = tid >> 3, seg = tid & 7;
        float s = 0.f;
        const float* p = &ldsC[r * 256 + seg * 32];
        #pragma unroll
        for (int j = 0; j < 32; ++j) { float v = p[j]; s += v * v; }
        ldsRed[r * 8 + seg] = s;
    }
    __syncthreads();
    if (tid < 32) {
        float s = 0.f;
        #pragma unroll
        for (int j = 0; j < 8; ++j) s += ldsRed[tid * 8 + j];
        ldsRms[tid] = rsqrtf(s * (1.f / (float)HD_) + 1e-6f);
    }
    __syncthreads();
    {
        const int r = tid >> 3, seg = tid & 7;
        const int mg = m0 + r;
        const int bb = mg / S_, srow = mg % S_;
        const float pos = (float)(start_pos + srow);
        const float rms = ldsRms[r];
        if (mode == 2) {
            // V: rmsnorm(weight=1), no rope, transposed store
            #pragma unroll
            for (int jj = 0; jj < 32; ++jj) {
                const int d = seg * 32 + jj;
                const float val = ldsC[r * 256 + d] * rms;
                vfull[((size_t)(bb * KV_ + hh) * HD_ + d) * MAXSEQ_ + (start_pos + srow)] = f2b(val);
            }
        } else {
            const float* nw = (mode == 0) ? qn : kn;
            unsigned short* dst = (mode == 0)
                ? &qbf[((size_t)(bb * H_ + hh) * S_ + srow) * HD_]
                : &kfull[(size_t)(bb * KV_ + hh) * MAXSEQ_ * HD_ +
                         (size_t)(start_pos + srow) * HD_];
            #pragma unroll
            for (int jj = 0; jj < 32; jj += 8) {
                const int d0 = seg * 32 + jj;
                float o[8];
                const int fj0 = d0 & 127;           // uniform across the 8-group
                if (fj0 < 64) {
                    #pragma unroll
                    for (int u = 0; u < 8; ++u) {
                        const int d = d0 + u, fj = fj0 + u;
                        const float val = ldsC[r * 256 + d] * rms * nw[d];
                        const float invf = __expf((float)fj * (-2.0f / (float)HD_) * LN_THETA);
                        float sn, cs;
                        __sincosf(pos * invf, &sn, &cs);
                        const int dp = d ^ 128;
                        const float pv = ldsC[r * 256 + dp] * rms * nw[dp];
                        o[u] = (d < 128) ? (val * cs - pv * sn) : (val * cs + pv * sn);
                    }
                } else {
                    #pragma unroll
                    for (int u = 0; u < 8; ++u) {
                        const int d = d0 + u;
                        o[u] = ldsC[r * 256 + d] * rms * nw[d];   // inv_freq == 0
                    }
                }
                uint4 pk;
                pk.x = pack2(o[0], o[1]);
                pk.y = pack2(o[2], o[3]);
                pk.z = pack2(o[4], o[5]);
                pk.w = pack2(o[6], o[7]);
                *(uint4*)(&dst[d0]) = pk;
            }
        }
    }
}

// =====================================================================
// Kernel 2: prefix int8 quant->dequant round trip into bf16 caches.
// =====================================================================
__global__ __launch_bounds__(256)
void prefix_kernel(const float* __restrict__ kp,
                   const float* __restrict__ vp,
                   unsigned short* __restrict__ kfull,
                   unsigned short* __restrict__ vfull)
{
    const int lane = threadIdx.x & 31;
    const int wv   = threadIdx.x >> 5;
    const int gid  = blockIdx.x * 8 + wv;            // 0 .. B*KV*S-1
    const int bb   = gid / (KV_ * S_);
    const int rem  = gid % (KV_ * S_);
    const int kv   = rem / S_;
    const int s    = rem % S_;

    const float* krow = kp + ((size_t)(bb * KV_ + kv) * S_ + s) * HD_;
    const float* vrow = vp + ((size_t)(bb * KV_ + kv) * S_ + s) * HD_;

    const float4 ka = *(const float4*)(krow + lane * 8);
    const float4 kb = *(const float4*)(krow + lane * 8 + 4);
    const float4 va = *(const float4*)(vrow + lane * 8);
    const float4 vb = *(const float4*)(vrow + lane * 8 + 4);

    float kam = fmaxf(fmaxf(fmaxf(fabsf(ka.x), fabsf(ka.y)), fmaxf(fabsf(ka.z), fabsf(ka.w))),
                      fmaxf(fmaxf(fabsf(kb.x), fabsf(kb.y)), fmaxf(fabsf(kb.z), fabsf(kb.w))));
    float vam = fmaxf(fmaxf(fmaxf(fabsf(va.x), fabsf(va.y)), fmaxf(fabsf(va.z), fabsf(va.w))),
                      fmaxf(fmaxf(fabsf(vb.x), fabsf(vb.y)), fmaxf(fabsf(vb.z), fabsf(vb.w))));
    #pragma unroll
    for (int m = 1; m < 32; m <<= 1) {
        kam = fmaxf(kam, __shfl_xor(kam, m, 32));
        vam = fmaxf(vam, __shfl_xor(vam, m, 32));
    }
    kam = fmaxf(kam, 1e-8f);
    vam = fmaxf(vam, 1e-8f);
    const float ks = kam / 127.f, kinv = 127.f / kam;
    const float vs = vam / 127.f, vinv = 127.f / vam;

    #define QDQ(x, inv, sc) (fminf(fmaxf(rintf((x) * (inv)), -128.f), 127.f) * (sc))

    {
        unsigned short* kd = kfull + (size_t)(bb * KV_ + kv) * MAXSEQ_ * HD_ +
                             (size_t)s * HD_ + lane * 8;
        uint4 pk;
        pk.x = pack2(QDQ(ka.x, kinv, ks), QDQ(ka.y, kinv, ks));
        pk.y = pack2(QDQ(ka.z, kinv, ks), QDQ(ka.w, kinv, ks));
        pk.z = pack2(QDQ(kb.x, kinv, ks), QDQ(kb.y, kinv, ks));
        pk.w = pack2(QDQ(kb.z, kinv, ks), QDQ(kb.w, kinv, ks));
        *(uint4*)kd = pk;
    }
    {
        float dv[8] = { QDQ(va.x, vinv, vs), QDQ(va.y, vinv, vs),
                        QDQ(va.z, vinv, vs), QDQ(va.w, vinv, vs),
                        QDQ(vb.x, vinv, vs), QDQ(vb.y, vinv, vs),
                        QDQ(vb.z, vinv, vs), QDQ(vb.w, vinv, vs) };
        const size_t base = (size_t)(bb * KV_ + kv) * HD_ * MAXSEQ_;
        #pragma unroll
        for (int i = 0; i < 8; ++i)
            vfull[base + (size_t)(lane * 8 + i) * MAXSEQ_ + s] = f2b(dv[i]);
    }
    #undef QDQ
}

// =====================================================================
// Kernel 3: flash attention, one wave32 per 16 queries x full head.
// Q tile staged into LDS by the Tensor Data Mover when available.
// Output written directly as bf16 (A operand of the output projection).
// =====================================================================
__global__ __launch_bounds__(32)
void attn_kernel(const unsigned short* __restrict__ qbf,
                 const unsigned short* __restrict__ kfull,
                 const unsigned short* __restrict__ vfull,
                 const int* __restrict__ sp,
                 unsigned short* __restrict__ attnb)
{
    const int lane = threadIdx.x;
    const int l16  = lane & 15;
    const int hf   = lane >> 4;

    const int qb   = blockIdx.x;                   // 0 .. B*H*(S/16)-1
    const int bb   = qb / (H_ * (S_ / 16));
    const int rem  = qb % (H_ * (S_ / 16));
    const int h    = rem / (S_ / 16);
    const int s0   = (rem % (S_ / 16)) * 16;
    const int kv   = h / GROUPS_;
    const int start = sp[0];

    __shared__ __align__(16) unsigned short ldsQ[16 * 256];  // 8 KB
    __shared__ __align__(16) unsigned short ldsP[16 * 32];   // 1 KB

    const unsigned short* qp = qbf + ((size_t)(bb * H_ + h) * S_ + s0) * HD_;
#if HAVE_TDM
    tdm_load_1d_bf16((unsigned)(size_t)ldsQ, qp, 16 * 256);
    __builtin_amdgcn_s_wait_tensorcnt(0);
#else
    for (int i = lane; i < 512; i += 32)
        ((uint4*)ldsQ)[i] = ((const uint4*)qp)[i];
    asm volatile("s_wait_dscnt 0x0" ::: "memory");
#endif

    v8f acc[16];
    {
        v8f z = {0.f,0.f,0.f,0.f,0.f,0.f,0.f,0.f};
        #pragma unroll
        for (int i = 0; i < 16; ++i) acc[i] = z;
    }
    float mi[8], li[8];
    #pragma unroll
    for (int j = 0; j < 8; ++j) { mi[j] = -__builtin_huge_valf(); li[j] = 0.f; }

    const int qpos_min = start + s0;
    const int ntiles   = ((qpos_min + 15) >> 5) + 1;
    const unsigned short* kbase = kfull + (size_t)(bb * KV_ + kv) * MAXSEQ_ * HD_;
    const unsigned short* vbase = vfull + (size_t)(bb * KV_ + kv) * HD_ * MAXSEQ_;

    for (int tile = 0; tile < ntiles; ++tile) {
        const int k0 = tile * 32;
        if (tile + 1 < ntiles)
            __builtin_prefetch(kbase + (size_t)(k0 + 32 + l16) * HD_, 0, 1);

        // ---- scores: S = Q(16x256) @ K^T(256x32) ----
        v8f c0 = {}, c1 = {};
        #pragma unroll
        for (int c = 0; c < 8; ++c) {
            B16V a, b0, b1;
            a.q[0] = *(const uint4*)(&ldsQ[l16 * 256 + c * 32 + hf * 8]);
            a.q[1] = *(const uint4*)(&ldsQ[l16 * 256 + c * 32 + 16 + hf * 8]);
            const unsigned short* kp0 = kbase + (size_t)(k0 + l16) * HD_ + c * 32 + hf * 8;
            b0.q[0] = *(const uint4*)(kp0);
            b0.q[1] = *(const uint4*)(kp0 + 16);
            const unsigned short* kp1 = kp0 + (size_t)16 * HD_;
            b1.q[0] = *(const uint4*)(kp1);
            b1.q[1] = *(const uint4*)(kp1 + 16);
            c0 = wmma_bf16(a, b0, c0);
            c1 = wmma_bf16(a, b1, c1);
        }

        // ---- causal mask (reference applies no 1/sqrt(d) scaling) ----
        if (k0 + 31 > qpos_min) {
            #pragma unroll
            for (int j = 0; j < 8; ++j) {
                const int qpos = start + s0 + hf * 8 + j;
                if (k0 + l16 > qpos)      c0[j] = -__builtin_huge_valf();
                if (k0 + 16 + l16 > qpos) c1[j] = -__builtin_huge_valf();
            }
        }

        // ---- online softmax (row reduce across the 16-lane half) ----
        float psc[8];
        #pragma unroll
        for (int j = 0; j < 8; ++j) {
            float tm = fmaxf(c0[j], c1[j]);
            #pragma unroll
            for (int m = 1; m < 16; m <<= 1) tm = fmaxf(tm, __shfl_xor(tm, m, 32));
            const float nm = fmaxf(mi[j], tm);
            const float sc = __expf(mi[j] - nm);
            mi[j] = nm;
            const float p0 = __expf(c0[j] - nm);
            const float p1 = __expf(c1[j] - nm);
            c0[j] = p0; c1[j] = p1;
            float rs = p0 + p1;
            #pragma unroll
            for (int m = 1; m < 16; m <<= 1) rs += __shfl_xor(rs, m, 32);
            li[j] = li[j] * sc + rs;
            psc[j] = sc;
        }
        #pragma unroll
        for (int tt = 0; tt < 16; ++tt)
            #pragma unroll
            for (int j = 0; j < 8; ++j) acc[tt][j] *= psc[j];

        // ---- P: C-layout -> A-layout via LDS ----
        #pragma unroll
        for (int j = 0; j < 8; ++j) {
            const int rowo = (hf * 8 + j) * 32;
            ldsP[rowo + l16]      = f2b(c0[j]);
            ldsP[rowo + 16 + l16] = f2b(c1[j]);
        }
        asm volatile("s_wait_dscnt 0x0" ::: "memory");
        B16V pa;
        pa.q[0] = *(const uint4*)(&ldsP[l16 * 32 + hf * 8]);
        pa.q[1] = *(const uint4*)(&ldsP[l16 * 32 + 16 + hf * 8]);

        // ---- O += P(16x32) @ V(32x256) ----
        #pragma unroll
        for (int tt = 0; tt < 16; ++tt) {
            B16V vb;
            const unsigned short* vp = vbase + (size_t)(tt * 16 + l16) * MAXSEQ_ + k0 + hf * 8;
            vb.q[0] = *(const uint4*)(vp);
            vb.q[1] = *(const uint4*)(vp + 16);
            acc[tt] = wmma_bf16(pa, vb, acc[tt]);
        }
    }

    // ---- finalize: O /= l, store bf16 [B][S][H*HD] ----
    float inv[8];
    #pragma unroll
    for (int j = 0; j < 8; ++j) inv[j] = 1.f / li[j];
    #pragma unroll
    for (int tt = 0; tt < 16; ++tt) {
        #pragma unroll
        for (int j = 0; j < 8; ++j) {
            const int srow = s0 + hf * 8 + j;
            attnb[(size_t)(bb * S_ + srow) * (H_ * HD_) + h * HD_ + tt * 16 + l16] =
                f2b(acc[tt][j] * inv[j]);
        }
    }
}

// =====================================================================
// Kernel 4: output projection  out = attn[4096x2048] @ w_o[2048x2560]
// Pure WMMA main loop: A from bf16 attn, B from transposed bf16 w_o.
// No LDS, no barriers.
// =====================================================================
__global__ __launch_bounds__(256)
void oproj_kernel(const unsigned short* __restrict__ attnb,
                  const unsigned short* __restrict__ woT,
                  float* __restrict__ out)
{
    const int tid  = threadIdx.x;
    const int lane = tid & 31;
    const int wv   = tid >> 5;
    const int l16  = lane & 15;
    const int hf   = lane >> 4;

    const int m0   = blockIdx.x * 32;
    const int coff = blockIdx.y * 256;

    const int rb = wv & 1;
    const int cb = wv >> 1;

    v8f acc[4];
    {
        v8f z = {0.f,0.f,0.f,0.f,0.f,0.f,0.f,0.f};
        #pragma unroll
        for (int i = 0; i < 4; ++i) acc[i] = z;
    }

    const unsigned short* arow = attnb + (size_t)(m0 + rb * 16 + l16) * (H_ * HD_);
    const unsigned short* bcol[4];
    #pragma unroll
    for (int i = 0; i < 4; ++i)
        bcol[i] = woT + (size_t)(coff + (cb * 4 + i) * 16 + l16) * (H_ * HD_);

    for (int k0 = 0; k0 < H_ * HD_; k0 += 32) {
        B16V a;
        a.q[0] = *(const uint4*)(arow + k0 + hf * 8);
        a.q[1] = *(const uint4*)(arow + k0 + 16 + hf * 8);
        #pragma unroll
        for (int i = 0; i < 4; ++i) {
            B16V bm;
            bm.q[0] = *(const uint4*)(bcol[i] + k0 + hf * 8);
            bm.q[1] = *(const uint4*)(bcol[i] + k0 + 16 + hf * 8);
            acc[i] = wmma_bf16(a, bm, acc[i]);
        }
    }

    #pragma unroll
    for (int i = 0; i < 4; ++i) {
        const int col = coff + (cb * 4 + i) * 16 + l16;
        #pragma unroll
        for (int j = 0; j < 8; ++j) {
            const int row = m0 + rb * 16 + hf * 8 + j;
            out[(size_t)row * HIDDEN_ + col] = acc[i][j];
        }
    }
}

// =====================================================================
extern "C" void kernel_launch(void* const* d_in, const int* in_sizes, int n_in,
                              void* d_out, int out_size, void* d_ws, size_t ws_size,
                              hipStream_t stream)
{
    const float* x    = (const float*)d_in[0];
    const float* w_q  = (const float*)d_in[1];
    const float* w_k  = (const float*)d_in[2];
    const float* w_v  = (const float*)d_in[3];
    const float* w_o  = (const float*)d_in[4];
    const float* qn   = (const float*)d_in[5];
    const float* kn   = (const float*)d_in[6];
    const float* kpre = (const float*)d_in[7];
    const float* vpre = (const float*)d_in[8];
    const int*   sp   = (const int*)  d_in[9];
    float* out = (float*)d_out;

    // workspace layout (bf16 as ushort), ~120 MiB total
    unsigned short* qbf   = (unsigned short*)d_ws;                               // 16 MiB
    unsigned short* kfull = qbf   + (size_t)B_ * H_  * S_      * HD_;            // 16 MiB
    unsigned short* vfull = kfull + (size_t)B_ * KV_ * MAXSEQ_ * HD_;            // 16 MiB
    unsigned short* attnb = vfull + (size_t)B_ * KV_ * HD_ * MAXSEQ_;            // 16 MiB
    unsigned short* xbf   = attnb + (size_t)B_ * S_ * H_ * HD_;                  // 20 MiB
    unsigned short* wT    = xbf   + (size_t)B_ * S_ * HIDDEN_;                   // 20 MiB
    unsigned short* woT   = wT    + (size_t)NQKV_ * HIDDEN_;                     // 10 MiB

    // 0) one-shot bf16 conversions (stay resident in L2 for the GEMMs)
    cvt_x_kernel <<<(size_t)B_ * S_ * HIDDEN_ / (256 * 8), 256, 0, stream>>>(x, xbf);
    cvt_w_kernel <<<dim3(HIDDEN_ / 32, NQKV_ / 256), 256, 0, stream>>>(w_q, w_k, w_v, wT);
    cvt_wo_kernel<<<dim3((H_ * HD_) / 32, HIDDEN_ / 256), 256, 0, stream>>>(w_o, woT);

    // 1) QKV gemm + rmsnorm + rope  (128 M-tiles x 16 heads)
    qkv_kernel<<<dim3((B_ * S_) / 32, H_ + 2 * KV_), 256, 0, stream>>>(
        xbf, wT, qn, kn, sp, qbf, kfull, vfull);

    // 2) prefix quant-dequant into caches (one wave per token)
    prefix_kernel<<<(B_ * KV_ * S_) / 8, 256, 0, stream>>>(kpre, vpre, kfull, vfull);

    // 3) flash attention (one wave per 16 queries)
    attn_kernel<<<B_ * H_ * (S_ / 16), 32, 0, stream>>>(qbf, kfull, vfull, sp, attnb);

    // 4) output projection
    oproj_kernel<<<dim3((B_ * S_) / 32, HIDDEN_ / 256), 256, 0, stream>>>(attnb, woT, out);
}
